// ContextSynthesizer_34875134444052
// MI455X (gfx1250) — matrix operand order repaired and checked by reading, not compiled, not run on hardware
//
#include <hip/hip_runtime.h>
#include <stdint.h>

#define NOTES_NUM 88
#define FREQ_BIN 1025
#define HALF 512
#define HID 512
#define Bq 4
#define Tq 512
#define T1 510   /* after conv1 */
#define TR 508   /* after conv2 */
#define NCOLS (Bq * TR)   /* 2032 */
#define SPEC_LD 1040      /* padded row stride for spec (16B-aligned) */

typedef __attribute__((ext_vector_type(16))) _Float16 v16h;
typedef __attribute__((ext_vector_type(8)))  _Float16 v8h;
typedef __attribute__((ext_vector_type(8)))  float    v8f;
typedef __attribute__((ext_vector_type(4)))  float    v4f;

// ---------------------------------------------------------------------------
// Kernel 1: repack weights.
//   W1  -> W1h[k][o][i]    (f16, per-tap contiguous K)
//   W2  -> W2h[o][i]       (f16)
//   W0  -> W0c[o][12]      (f32, conv1 collapsed to 4 streams x 3 taps)
// ---------------------------------------------------------------------------
__global__ void repack_kernel(const float* __restrict__ W1, const float* __restrict__ W2,
                              const float* __restrict__ W0, const float* __restrict__ Wf0,
                              _Float16* __restrict__ W1h, _Float16* __restrict__ W2h,
                              float* __restrict__ W0c) {
  const int idx = blockIdx.x * 256 + threadIdx.x;
  if (idx < 3 * HID * HID) {
    const int i = idx & 511;
    const int rest = idx >> 9;
    const int o = rest & 511;
    const int k = rest >> 9;
    W1h[idx] = (_Float16)W1[((size_t)o * HID + i) * 3 + k];
  }
  if (idx < HID * HID) W2h[idx] = (_Float16)W2[idx];
  if (idx < HID) {
    const int o = idx;
    const float* w = W0 + (size_t)o * 19 * 3;
    #pragma unroll
    for (int k = 0; k < 3; ++k) {
      W0c[o * 12 + 0 * 3 + k] = w[0 * 3 + k];
      float acc = 0.f;
      #pragma unroll
      for (int j = 0; j < 16; ++j) acc += w[(1 + j) * 3 + k] * Wf0[j];
      W0c[o * 12 + 1 * 3 + k] = acc;
      W0c[o * 12 + 2 * 3 + k] = w[17 * 3 + k];
      W0c[o * 12 + 3 * 3 + k] = w[18 * 3 + k];
    }
  }
}

// ---------------------------------------------------------------------------
// Kernel 2: conv1 + relu -> h1t[b*T1 + t][i]  (t-major, f16)
// grid (T1, B), block 256 over output channels (2 each); coalesced stores.
// ---------------------------------------------------------------------------
__global__ void conv1_kernel(const float* __restrict__ ru, const float* __restrict__ f0,
                             const float* __restrict__ nf, const float* __restrict__ W0c,
                             const float* __restrict__ b0, _Float16* __restrict__ h1t) {
  const int t = blockIdx.x;
  const int b = blockIdx.y;
  const int tid = threadIdx.x;
  __shared__ float pv[12];            // pv[s*3+k] = stream s at tap t+k
  if (tid < 12) {
    const int s = tid / 3, k = tid - s * 3;
    const int tau = t + k;
    float v;
    if (s == 0)      v = ru[(size_t)b * Tq + tau];
    else if (s == 1) v = f0[(size_t)b * Tq + tau];
    else if (s == 2) v = nf[(size_t)(b * 2) * Tq + tau];
    else             v = nf[(size_t)(b * 2 + 1) * Tq + tau];
    pv[tid] = v;
  }
  __syncthreads();
  _Float16* orow = h1t + ((size_t)b * T1 + t) * HID;
  for (int o = tid; o < HID; o += 256) {
    const float* cw = W0c + o * 12;
    float s = b0[o];
    #pragma unroll
    for (int j = 0; j < 12; ++j) s += cw[j] * pv[j];
    s = s > 0.f ? s : 0.f;
    orow[o] = (_Float16)s;
  }
}

// ---------------------------------------------------------------------------
// WMMA helper: C += A(16xK) * B(Kx16), both operands contiguous-K per lane.
// Arow: base of lane's A row.  Bcol: base of lane's B column (t-major act).
// ---------------------------------------------------------------------------
__device__ __forceinline__ v8f wmma_span(const _Float16* __restrict__ Arow,
                                         const _Float16* __restrict__ Bcol,
                                         int K, int hi, v8f c) {
  __builtin_prefetch(Arow, 0, 0);
  __builtin_prefetch(Bcol, 0, 0);
  for (int k = 0; k < K; k += 32) {
    v8h alo = *(const v8h*)(Arow + k + hi * 8);
    v8h ahi = *(const v8h*)(Arow + k + 16 + hi * 8);
    v8h blo = *(const v8h*)(Bcol + k + hi * 8);
    v8h bhi = *(const v8h*)(Bcol + k + 16 + hi * 8);
    v16h a = __builtin_shufflevector(alo, ahi, 0, 1, 2, 3, 4, 5, 6, 7,
                                     8, 9, 10, 11, 12, 13, 14, 15);
    v16h b = __builtin_shufflevector(blo, bhi, 0, 1, 2, 3, 4, 5, 6, 7,
                                     8, 9, 10, 11, 12, 13, 14, 15);
    c = __builtin_amdgcn_wmma_f32_16x16x32_f16(false, a, false, b, (short)0, c, false, false);
  }
  return c;
}

// ---------------------------------------------------------------------------
// Kernel 3: conv2 = 3 accumulating GEMMs (M=512, N=508, K=512) + relu
// grid (8, 16, B), block 256 (8 waves, 64x32 block tile); h2t t-major f16.
// ---------------------------------------------------------------------------
__global__ void conv2_wmma(const _Float16* __restrict__ W1h, const _Float16* __restrict__ h1t,
                           const float* __restrict__ b1, _Float16* __restrict__ h2t) {
  const int tid = threadIdx.x;
  const int l  = tid & 15;
  const int hi = (tid >> 4) & 1;
  const int wv = tid >> 5;
  const int bb = blockIdx.z;
  const int M0 = blockIdx.x * 64 + (wv >> 1) * 16;
  const int N0 = blockIdx.y * 32 + (wv & 1) * 16;
  int n = N0 + l; if (n > TR - 1) n = TR - 1;
  v8f c = {};
  #pragma unroll
  for (int kk = 0; kk < 3; ++kk) {
    const _Float16* Arow = W1h + (size_t)kk * HID * HID + (size_t)(M0 + l) * HID;
    const _Float16* Bcol = h1t + ((size_t)bb * T1 + n + kk) * HID;
    c = wmma_span(Arow, Bcol, HID, hi, c);
  }
  const int nOut = N0 + l;
  if (nOut < TR) {
    const int o0 = M0 + hi * 8;
    v8h hv;
    #pragma unroll
    for (int r = 0; r < 8; ++r) {
      float v = c[r] + b1[o0 + r];
      hv[r] = (_Float16)(v > 0.f ? v : 0.f);
    }
    *(v8h*)(h2t + ((size_t)bb * TR + nOut) * HID + o0) = hv;
  }
}

// ---------------------------------------------------------------------------
// Kernel 4: rep = W2 @ h2 + b2 (M=512, N=508, K=512); rept t-major f16
// ---------------------------------------------------------------------------
__global__ void rep_wmma(const _Float16* __restrict__ W2h, const _Float16* __restrict__ h2t,
                         const float* __restrict__ b2, _Float16* __restrict__ rept) {
  const int tid = threadIdx.x;
  const int l  = tid & 15;
  const int hi = (tid >> 4) & 1;
  const int wv = tid >> 5;
  const int bb = blockIdx.z;
  const int M0 = blockIdx.x * 64 + (wv >> 1) * 16;
  const int N0 = blockIdx.y * 32 + (wv & 1) * 16;
  int n = N0 + l; if (n > TR - 1) n = TR - 1;
  const _Float16* Arow = W2h + (size_t)(M0 + l) * HID;
  const _Float16* Bcol = h2t + ((size_t)bb * TR + n) * HID;
  v8f c = wmma_span(Arow, Bcol, HID, hi, v8f{});
  const int nOut = N0 + l;
  if (nOut < TR) {
    const int o0 = M0 + hi * 8;
    v8h hv;
    #pragma unroll
    for (int r = 0; r < 8; ++r) hv[r] = (_Float16)(c[r] + b2[o0 + r]);
    *(v8h*)(rept + ((size_t)bb * TR + nOut) * HID + o0) = hv;
  }
}

// ---------------------------------------------------------------------------
// Kernel 5: bucket (b,t) columns by note. One block, 256 threads.
// ---------------------------------------------------------------------------
__global__ void group_kernel(const int* __restrict__ note_idx, int* __restrict__ gstart,
                             int* __restrict__ gcols) {
  __shared__ int cnt[NOTES_NUM];
  __shared__ int off[NOTES_NUM];
  const int tid = threadIdx.x;
  if (tid < NOTES_NUM) cnt[tid] = 0;
  __syncthreads();
  for (int col = tid; col < NCOLS; col += 256) {
    const int b = col / TR, t = col - b * TR;
    atomicAdd(&cnt[note_idx[b * Tq + t + 2]], 1);
  }
  __syncthreads();
  if (tid == 0) {
    int run = 0;
    for (int n = 0; n < NOTES_NUM; ++n) { off[n] = run; gstart[n] = run; run += cnt[n]; }
    gstart[NOTES_NUM] = run;
  }
  __syncthreads();
  for (int col = tid; col < NCOLS; col += 256) {
    const int b = col / TR, t = col - b * TR;
    const int p = atomicAdd(&off[note_idx[b * Tq + t + 2]], 1);
    gcols[p] = col;
  }
}

// ---------------------------------------------------------------------------
// Kernel 6: ori_en[b,t] = sum_c first_frame[b,c,t+2]
// ---------------------------------------------------------------------------
__global__ void energy_kernel(const float* __restrict__ ff, float* __restrict__ en_ws) {
  const int b = blockIdx.y;
  const int t = blockIdx.x * 256 + threadIdx.x;
  if (t >= TR) return;
  const float* p = ff + (size_t)b * FREQ_BIN * Tq + t + 2;
  float s = 0.f;
  for (int c = 0; c < FREQ_BIN; ++c) s += p[(size_t)c * Tq];
  en_ws[b * TR + t] = s;
}

// ---------------------------------------------------------------------------
// Kernel 7: per-note GEMM  spec[cols_of_note][:] = W_pitch[n] @ rep[:,cols]
// grid (65 M-tiles, 16 N-tiles, 88 notes), block 32 (one wave per 16x16 tile)
// A = W_pitch rows (f32, cvt to f16 after b128 loads); B = rept (t-major).
// ---------------------------------------------------------------------------
__global__ void spec_wmma(const float* __restrict__ Wp, const _Float16* __restrict__ rept,
                          const int* __restrict__ gstart, const int* __restrict__ gcols,
                          float* __restrict__ spec, const int* __restrict__ lowp,
                          const int* __restrict__ upp) {
  const int n = blockIdx.z;
  const int low = *lowp, up = *upp;
  const bool active = (n >= low && n < up) || (n == NOTES_NUM - 1);
  if (!active) return;
  const int s0 = gstart[n];
  const int count = gstart[n + 1] - s0;
  const int N0 = blockIdx.y * 16;
  if (N0 >= count) return;
  const int M0 = blockIdx.x * 16;
  const int tid = threadIdx.x;
  const int l  = tid & 15;
  const int hi = (tid >> 4) & 1;
  const int j = N0 + l;
  const int jc = j < count ? j : count - 1;
  const int col = gcols[s0 + jc];
  const _Float16* Bcol = rept + (size_t)col * HID;
  int o = M0 + l; if (o > FREQ_BIN - 1) o = FREQ_BIN - 1;
  const float* Arow = Wp + ((size_t)n * FREQ_BIN + o) * HALF;
  __builtin_prefetch(Arow, 0, 0);
  v8f c = {};
  for (int k = 0; k < HALF; k += 32) {
    v4f a0 = *(const v4f*)(Arow + k + hi * 8);
    v4f a1 = *(const v4f*)(Arow + k + hi * 8 + 4);
    v4f a2 = *(const v4f*)(Arow + k + 16 + hi * 8);
    v4f a3 = *(const v4f*)(Arow + k + 16 + hi * 8 + 4);
    v16h a;
    #pragma unroll
    for (int q = 0; q < 4; ++q) {
      a[q]      = (_Float16)a0[q];
      a[4 + q]  = (_Float16)a1[q];
      a[8 + q]  = (_Float16)a2[q];
      a[12 + q] = (_Float16)a3[q];
    }
    v8h blo = *(const v8h*)(Bcol + k + hi * 8);
    v8h bhi = *(const v8h*)(Bcol + k + 16 + hi * 8);
    v16h b = __builtin_shufflevector(blo, bhi, 0, 1, 2, 3, 4, 5, 6, 7,
                                     8, 9, 10, 11, 12, 13, 14, 15);
    c = __builtin_amdgcn_wmma_f32_16x16x32_f16(false, a, false, b, (short)0, c, false, false);
  }
  if (j < count) {
    // lane holds 8 consecutive o at fixed col; two aligned 16B stores
    float* op = spec + (size_t)col * SPEC_LD + M0 + hi * 8;
    v4f lo, hs;
    #pragma unroll
    for (int r = 0; r < 4; ++r) { lo[r] = c[r]; hs[r] = c[4 + r]; }
    *(v4f*)(op)     = lo;
    *(v4f*)(op + 4) = hs;
  }
}

// ---------------------------------------------------------------------------
// Kernel 8: per-column softmax over 1025 bins * ori_en; inactive -> uniform
// grid 2032, block 256
// ---------------------------------------------------------------------------
__global__ void softmax_kernel(const float* __restrict__ spec, const float* __restrict__ en_ws,
                               const int* __restrict__ note_idx, float* __restrict__ out,
                               const int* __restrict__ lowp, const int* __restrict__ upp) {
  const int col = blockIdx.x;
  const int bb = col / TR, tt = col - bb * TR;
  const int n = note_idx[bb * Tq + tt + 2];
  const int low = *lowp, up = *upp;
  const float en = en_ws[col];
  float* outc = out + (size_t)bb * FREQ_BIN * TR + tt;   // element o at outc[o*TR]
  const int tid = threadIdx.x;
  __shared__ float red[256];
  const bool active = (n >= low && n < up) || (n == NOTES_NUM - 1);
  if (!active) {
    const float v = en * (1.0f / 1025.0f);
    for (int o = tid; o < FREQ_BIN; o += 256) outc[(size_t)o * TR] = v;
    return;
  }
  const float* sc = spec + (size_t)col * SPEC_LD;
  float m = -3.4e38f;
  for (int o = tid; o < FREQ_BIN; o += 256) m = fmaxf(m, sc[o]);
  red[tid] = m; __syncthreads();
  for (int s = 128; s > 0; s >>= 1) { if (tid < s) red[tid] = fmaxf(red[tid], red[tid + s]); __syncthreads(); }
  m = red[0]; __syncthreads();
  float ssum = 0.f;
  for (int o = tid; o < FREQ_BIN; o += 256) ssum += __expf(sc[o] - m);
  red[tid] = ssum; __syncthreads();
  for (int s = 128; s > 0; s >>= 1) { if (tid < s) red[tid] += red[tid + s]; __syncthreads(); }
  const float inv = en / red[0];
  for (int o = tid; o < FREQ_BIN; o += 256) outc[(size_t)o * TR] = __expf(sc[o] - m) * inv;
}

// ---------------------------------------------------------------------------
extern "C" void kernel_launch(void* const* d_in, const int* in_sizes, int n_in,
                              void* d_out, int out_size, void* d_ws, size_t ws_size,
                              hipStream_t stream) {
  (void)in_sizes; (void)n_in; (void)out_size; (void)ws_size;
  const float* ff  = (const float*)d_in[0];
  const float* f0w = (const float*)d_in[1];
  const float* nf  = (const float*)d_in[2];
  const float* ru  = (const float*)d_in[3];
  const float* Wf0 = (const float*)d_in[4];
  const float* W0  = (const float*)d_in[5];
  const float* b0  = (const float*)d_in[6];
  const float* W1  = (const float*)d_in[7];
  const float* b1  = (const float*)d_in[8];
  const float* W2  = (const float*)d_in[9];
  const float* b2  = (const float*)d_in[10];
  const float* Wp  = (const float*)d_in[11];
  const int* nidx  = (const int*)d_in[12];
  const int* lowp  = (const int*)d_in[13];
  const int* upp   = (const int*)d_in[14];
  float* out = (float*)d_out;

  char* p = (char*)d_ws;
  auto carve = [&](size_t bytes) -> char* {
    char* q = (char*)(((uintptr_t)p + 255) & ~(uintptr_t)255);
    p = q + bytes;
    return q;
  };
  _Float16* h1t  = (_Float16*)carve((size_t)Bq * T1 * HID * 2);
  _Float16* W1h  = (_Float16*)carve((size_t)3 * HID * HID * 2);
  _Float16* W2h  = (_Float16*)carve((size_t)HID * HID * 2);
  float*    W0c  = (float*)carve((size_t)HID * 12 * 4);
  _Float16* h2t  = (_Float16*)carve((size_t)Bq * TR * HID * 2);
  _Float16* rept = (_Float16*)carve((size_t)Bq * TR * HID * 2);
  float* spec    = (float*)carve((size_t)NCOLS * SPEC_LD * 4);
  float* en_ws   = (float*)carve((size_t)NCOLS * 4);
  int* gstart    = (int*)carve((NOTES_NUM + 1) * 4);
  int* gcols     = (int*)carve((size_t)NCOLS * 4);

  repack_kernel<<<dim3((3 * HID * HID + 255) / 256), 256, 0, stream>>>(W1, W2, W0, Wf0,
                                                                       W1h, W2h, W0c);
  conv1_kernel<<<dim3(T1, Bq), 256, 0, stream>>>(ru, f0w, nf, W0c, b0, h1t);
  conv2_wmma<<<dim3(HID / 64, (TR + 31) / 32, Bq), 256, 0, stream>>>(W1h, h1t, b1, h2t);
  rep_wmma<<<dim3(HID / 64, (TR + 31) / 32, Bq), 256, 0, stream>>>(W2h, h2t, b2, rept);
  group_kernel<<<1, 256, 0, stream>>>(nidx, gstart, gcols);
  energy_kernel<<<dim3((TR + 255) / 256, Bq), 256, 0, stream>>>(ff, en_ws);
  spec_wmma<<<dim3((FREQ_BIN + 15) / 16, 16, NOTES_NUM), 32, 0, stream>>>(Wp, rept, gstart, gcols,
                                                                          spec, lowp, upp);
  softmax_kernel<<<dim3(NCOLS), 256, 0, stream>>>(spec, en_ws, nidx, out, lowp, upp);
}
